// GeniePathLayer_6408091205710
// MI455X (gfx1250) — compile-verified, matrix operand order checked by reading, not run on hardware
//
#include <hip/hip_runtime.h>
#include <hip/hip_bf16.h>
#include <math.h>

// ---------------------------------------------------------------------------
// GeniePathLayer: GATConv (softmax aggregation) + 1-step LSTM, N=50000,
// E=600000, D=H=128.  bf16 WMMA (f32 accumulate) for all GEMMs; weights are
// pre-packed into per-lane WMMA B-fragment layout; each wave owns TWO 16-row
// A tiles so every B-fragment load feeds two v_wmma ops.
// __launch_bounds__(256,1) lifts the VGPR cap so accumulators never spill.
// Outputs: [h1, h1, c1] flattened.
// ---------------------------------------------------------------------------

typedef __attribute__((ext_vector_type(16))) __bf16       v16bf;
typedef __attribute__((ext_vector_type(8)))  float        v8f;
typedef __attribute__((ext_vector_type(8)))  unsigned int v8u;

#define DD 128           // feature dim D == H
#define NEG_SLOPE 0.2f

__device__ inline v8f wmma_bf16(v16bf a, v16bf b, v8f c) {
  return __builtin_amdgcn_wmma_f32_16x16x32_bf16(false, a, false, b,
                                                 (short)0, c, false, false);
}

__device__ inline v16bf as_v16bf(v8u r) { return __builtin_bit_cast(v16bf, r); }

__device__ inline float sigmoidf_(float x) { return 1.0f / (1.0f + __expf(-x)); }

// float -> bf16 bits, round to nearest even
__device__ inline unsigned short f2bf(float f) {
  unsigned u = __float_as_uint(f);
  u += 0x7FFFu + ((u >> 16) & 1u);
  return (unsigned short)(u >> 16);
}

// atomic max on float via signed/unsigned int atomics (valid for all orderings)
__device__ inline void atomicMaxF(float* addr, float val) {
  if (val >= 0.0f) atomicMax((int*)addr, __float_as_int(val));
  else             atomicMin((unsigned int*)addr, (unsigned int)__float_as_int(val));
}

// pack 16 floats (two groups of 8) into a bf16 A-fragment (hardware cvt)
__device__ inline v16bf pack_frag(const float* g0, const float* g1) {
  v16bf v;
  float4 a = *(const float4*)(g0);
  float4 b = *(const float4*)(g0 + 4);
  float4 c = *(const float4*)(g1);
  float4 d = *(const float4*)(g1 + 4);
  v[0]=(__bf16)a.x;  v[1]=(__bf16)a.y;  v[2]=(__bf16)a.z;  v[3]=(__bf16)a.w;
  v[4]=(__bf16)b.x;  v[5]=(__bf16)b.y;  v[6]=(__bf16)b.z;  v[7]=(__bf16)b.w;
  v[8]=(__bf16)c.x;  v[9]=(__bf16)c.y;  v[10]=(__bf16)c.z; v[11]=(__bf16)c.w;
  v[12]=(__bf16)d.x; v[13]=(__bf16)d.y; v[14]=(__bf16)d.z; v[15]=(__bf16)d.w;
  return v;
}

// ---------------------------------------------------------------------------
// Prep kernel: pack a K=128 x ncols B-matrix into per-lane WMMA fragments.
//   transposed==0:  B[k][n] = src[k*ncols + n]       (W, [128,128] row-major)
//   transposed==1:  B[k][n] = src[n*128 + k]         (W_ih/W_hh^T, src [ncols,128])
// Fragment f = kt*(ncols/16) + t ; lane l holds 16 bf16 (32 B) at dst + (f*32+l)*16.
// ---------------------------------------------------------------------------
__global__ void k_pack_b(const float* __restrict__ src, unsigned short* __restrict__ dst,
                         int ncols, int transposed) {
  const int tid   = blockIdx.x * 256 + threadIdx.x;
  const int tmax  = ncols >> 4;
  const int nfrag = 4 * tmax;
  if (tid >= nfrag * 32) return;
  const int lane = tid & 31;
  const int f    = tid >> 5;
  const int t    = f % tmax;
  const int kt   = f / tmax;
  const int nc   = t * 16 + (lane & 15);
  const int kb   = kt * 32 + (lane >> 4) * 8;
  unsigned short* dp = dst + (size_t)tid * 16;
  if (transposed) {
    const float* sp = src + (size_t)nc * DD;
#pragma unroll
    for (int j = 0; j < 8; ++j) {
      dp[j]     = f2bf(sp[kb + j]);
      dp[j + 8] = f2bf(sp[kb + 16 + j]);
    }
  } else {
#pragma unroll
    for (int j = 0; j < 8; ++j) {
      dp[j]     = f2bf(src[(size_t)(kb + j) * ncols + nc]);
      dp[j + 8] = f2bf(src[(size_t)(kb + 16 + j) * ncols + nc]);
    }
  }
}

// ---------------------------------------------------------------------------
// Kernel 1: xs = x @ W.  8 waves/block; each wave owns 32 rows (two 16-row
// tiles) so every B fragment is loaded once and used by two WMMAs.
// ---------------------------------------------------------------------------
__global__ void __launch_bounds__(256, 1)
k_gemm_xs(const float* __restrict__ x,
          const unsigned short* __restrict__ packW,
          float* __restrict__ xs, int n) {
  const int lane  = threadIdx.x & 31;
  const int wave  = threadIdx.x >> 5;
  const int row0  = (blockIdx.x * 8 + wave) * 32;
  if (row0 >= n) return;

  const int m     = lane & 15;
  const int khalf = (lane >> 4) * 8;
  const float* xrow[2];
#pragma unroll
  for (int tt = 0; tt < 2; ++tt) {
    int arow = row0 + tt * 16 + m; if (arow >= n) arow = n - 1;
    xrow[tt] = x + (size_t)arow * DD;
    __builtin_prefetch(xrow[tt], 0, 3);      // global_prefetch_b8
  }

  v8f acc[2][8];
  const v8f Z = {0.f,0.f,0.f,0.f,0.f,0.f,0.f,0.f};
#pragma unroll
  for (int tt = 0; tt < 2; ++tt)
#pragma unroll
    for (int t = 0; t < 8; ++t) acc[tt][t] = Z;

#pragma unroll
  for (int kt = 0; kt < 4; ++kt) {
    const int k0 = kt * 32;
    v16bf a0 = pack_frag(xrow[0] + k0 + khalf, xrow[0] + k0 + 16 + khalf);
    v16bf a1 = pack_frag(xrow[1] + k0 + khalf, xrow[1] + k0 + 16 + khalf);
#pragma unroll
    for (int t = 0; t < 8; ++t) {
      const v8u braw = *(const v8u*)(packW + ((size_t)((kt * 8 + t) * 32 + lane)) * 16);
      const v16bf b = as_v16bf(braw);
      acc[0][t] = wmma_bf16(a0, b, acc[0][t]);
      acc[1][t] = wmma_bf16(a1, b, acc[1][t]);
    }
  }
#pragma unroll
  for (int tt = 0; tt < 2; ++tt) {
#pragma unroll
    for (int t = 0; t < 8; ++t) {
      const int nc = t * 16 + m;
#pragma unroll
      for (int r = 0; r < 8; ++r) {
        const int M = row0 + tt * 16 + r + (lane >> 4) * 8;
        if (M < n) xs[(size_t)M * DD + nc] = acc[tt][t][r];
      }
    }
  }
}

// ---------------------------------------------------------------------------
// Kernel 2: per-node attention scalars; one wave per node, shuffle reduction
// ---------------------------------------------------------------------------
__global__ void k_att(const float* __restrict__ xs,
                      const float* __restrict__ att_s, const float* __restrict__ att_d,
                      float* __restrict__ a_src, float* __restrict__ a_dst, int n) {
  const int lane = threadIdx.x & 31;
  const int node = blockIdx.x * 8 + (threadIdx.x >> 5);
  if (node >= n) return;
  float4 v  = ((const float4*)(xs + (size_t)node * DD))[lane];
  float4 s4 = ((const float4*)att_s)[lane];
  float4 d4 = ((const float4*)att_d)[lane];
  float vs = v.x*s4.x + v.y*s4.y + v.z*s4.z + v.w*s4.w;
  float vd = v.x*d4.x + v.y*d4.y + v.z*d4.z + v.w*d4.w;
#pragma unroll
  for (int off = 16; off > 0; off >>= 1) {
    vs += __shfl_xor(vs, off, 32);
    vd += __shfl_xor(vd, off, 32);
  }
  if (lane == 0) { a_src[node] = vs; a_dst[node] = vd; }
}

// ---------------------------------------------------------------------------
// Kernel 3: init accumulators (accum=0, e_max=-inf, e_sum=0)
// ---------------------------------------------------------------------------
__global__ void k_init(float* __restrict__ accum, float* __restrict__ emax,
                       float* __restrict__ esum, int n) {
  const int t = blockIdx.x * 256 + threadIdx.x;
  if (t < n * DD) accum[t] = 0.0f;
  if (t < n) { emax[t] = __int_as_float(0xFF800000); esum[t] = 0.0f; }
}

// ---------------------------------------------------------------------------
// Kernel 4: edge pass 1 — e = leaky_relu(a_src[s]+a_dst[d]); segment max
// ---------------------------------------------------------------------------
__global__ void k_edge_max(const long long* __restrict__ ei,
                           const float* __restrict__ a_src, const float* __restrict__ a_dst,
                           float* __restrict__ ev, float* __restrict__ emax,
                           int Etot, int E) {
  const int t = blockIdx.x * 256 + threadIdx.x;
  if (t >= Etot) return;
  int s, d;
  if (t < E) { s = (int)ei[t]; d = (int)ei[(size_t)E + t]; }
  else       { s = d = t - E; }
  float e = a_src[s] + a_dst[d];
  e = (e >= 0.0f) ? e : NEG_SLOPE * e;
  ev[t] = e;
  atomicMaxF(&emax[d], e);
}

// ---------------------------------------------------------------------------
// Kernel 5: edge pass 2 — w = exp(e - max); esum[d]+=w; accum[d] += w*xs[s]
// one wave per edge: lane covers 4 of the 128 features
// ---------------------------------------------------------------------------
__global__ void k_edge_scatter(const long long* __restrict__ ei,
                               const float* __restrict__ ev, const float* __restrict__ emax,
                               const float* __restrict__ xs,
                               float* __restrict__ esum, float* __restrict__ accum,
                               int Etot, int E) {
  const int lane = threadIdx.x & 31;
  const int eidx = blockIdx.x * 8 + (threadIdx.x >> 5);
  if (eidx >= Etot) return;
  int s, d;
  if (eidx < E) { s = (int)ei[eidx]; d = (int)ei[(size_t)E + eidx]; }
  else          { s = d = eidx - E; }
  const float w = __expf(ev[eidx] - emax[d]);
  if (lane == 0) atomicAdd(&esum[d], w);
  const float4 v = ((const float4*)(xs + (size_t)s * DD))[lane];
  float* dst = accum + (size_t)d * DD + lane * 4;
  atomicAdd(dst + 0, w * v.x);
  atomicAdd(dst + 1, w * v.y);
  atomicAdd(dst + 2, w * v.z);
  atomicAdd(dst + 3, w * v.w);
}

// ---------------------------------------------------------------------------
// Kernel 6: fused normalize + bias + tanh -> xb (stored in LDS already packed
// as bf16 A-fragments), then LSTM gates via WMMA from pre-packed W_ih/W_hh.
// Two 16-row tiles per wave: each B-fragment load pair feeds 2 WMMAs.
// out = [h1 (n*128), h1 (n*128), c1 (n*128)]
// ---------------------------------------------------------------------------
__global__ void __launch_bounds__(256, 1)
k_lstm(const float* __restrict__ accum, const float* __restrict__ esum,
       const float* __restrict__ bias,
       const float* __restrict__ h0, const float* __restrict__ c0,
       const unsigned short* __restrict__ packIh,
       const unsigned short* __restrict__ packHh,
       float* __restrict__ out, int n) {
  // per-wave packed xb tiles: 2 x (16 rows x 128 k) bf16 A-fragments (8 KB/wave)
  __shared__ __align__(32) unsigned short xbp[8][2 * 16 * DD];
  const int lane = threadIdx.x & 31;
  const int wave = threadIdx.x >> 5;
  const int row0 = (blockIdx.x * 8 + wave) * 32;
  if (row0 >= n) return;
  unsigned short* xw = xbp[wave];

  // xb tiles: tanh(accum/(esum+eps) + bias), stored directly in fragment layout.
  // Pair (k, k+1), k even, lands in the same lane/element-pair -> one b32 store.
  for (int idx = lane; idx < 2 * 16 * 64; idx += 32) {
    const int tt = idx >> 10;          // which 16-row tile
    const int r  = (idx >> 6) & 15;    // row in tile
    const int kp = (idx & 63) * 2;     // even k
    int M = row0 + tt * 16 + r; if (M >= n) M = n - 1;
    const float den = esum[M] + 1e-16f;
    const float* ap = accum + (size_t)M * DD + kp;
    const float v0 = tanhf(ap[0] / den + bias[kp]);
    const float v1 = tanhf(ap[1] / den + bias[kp + 1]);
    const int kt = kp >> 5, rr = kp & 31;
    const int half = (rr & 8) ? 1 : 0;
    const int j    = (rr & 7) + ((rr & 16) ? 8 : 0);
    const int l    = half * 16 + r;
    const unsigned int pair = (unsigned)f2bf(v0) | ((unsigned)f2bf(v1) << 16);
    *(unsigned int*)(xw + tt * (16 * DD) + ((kt * 32 + l) * 16 + j)) = pair;
  }
  // make cross-lane LDS writes visible before fragment reads (same wave)
  asm volatile("s_wait_dscnt 0x0" ::: "memory");

  const int m     = lane & 15;
  const int khalf = (lane >> 4) * 8;
  const float* hrp[2];
#pragma unroll
  for (int tt = 0; tt < 2; ++tt) {
    int hrow = row0 + tt * 16 + m; if (hrow >= n) hrow = n - 1;
    hrp[tt] = h0 + (size_t)hrow * DD;
    __builtin_prefetch(hrp[tt], 0, 3);
  }

  // hoisted A fragments (reused by all 8 column tiles)
  v16bf a_x[4][2], a_h[4][2];
#pragma unroll
  for (int kt = 0; kt < 4; ++kt) {
#pragma unroll
    for (int tt = 0; tt < 2; ++tt) {
      a_x[kt][tt] = as_v16bf(*(const v8u*)(xw + tt * (16 * DD) + (kt * 32 + lane) * 16));
      a_h[kt][tt] = pack_frag(hrp[tt] + kt * 32 + khalf, hrp[tt] + kt * 32 + 16 + khalf);
    }
  }

  const v8f Z = {0.f,0.f,0.f,0.f,0.f,0.f,0.f,0.f};
  for (int t = 0; t < 8; ++t) {
    v8f g_acc[4][2];
#pragma unroll
    for (int g = 0; g < 4; ++g) { g_acc[g][0] = Z; g_acc[g][1] = Z; }
#pragma unroll
    for (int kt = 0; kt < 4; ++kt) {
#pragma unroll
      for (int gate = 0; gate < 4; ++gate) {
        const int f = kt * 32 + gate * 8 + t;   // fragment id (ncols=512)
        const v16bf bi = as_v16bf(*(const v8u*)(packIh + ((size_t)(f * 32 + lane)) * 16));
        g_acc[gate][0] = wmma_bf16(a_x[kt][0], bi, g_acc[gate][0]);
        g_acc[gate][1] = wmma_bf16(a_x[kt][1], bi, g_acc[gate][1]);
        const v16bf bh = as_v16bf(*(const v8u*)(packHh + ((size_t)(f * 32 + lane)) * 16));
        g_acc[gate][0] = wmma_bf16(a_h[kt][0], bh, g_acc[gate][0]);
        g_acc[gate][1] = wmma_bf16(a_h[kt][1], bh, g_acc[gate][1]);
      }
    }
    // epilogue: gate nonlinearities + state update (torch order i,f,g,o)
#pragma unroll
    for (int tt = 0; tt < 2; ++tt) {
#pragma unroll
      for (int r = 0; r < 8; ++r) {
        const int M = row0 + tt * 16 + r + (lane >> 4) * 8;
        if (M < n) {
          const int nc = t * 16 + m;
          const float iv = sigmoidf_(g_acc[0][tt][r]);
          const float fv = sigmoidf_(g_acc[1][tt][r]);
          const float gv = tanhf(g_acc[2][tt][r]);
          const float ov = sigmoidf_(g_acc[3][tt][r]);
          const float c0v = c0[(size_t)M * DD + nc];
          const float c1v = fv * c0v + iv * gv;
          const float h1v = ov * tanhf(c1v);
          const size_t o  = (size_t)M * DD + nc;
          const size_t NH = (size_t)n * DD;
          out[o]          = h1v;  // output #1 (xb*0 + h1 == h1)
          out[NH + o]     = h1v;  // h1[None]
          out[2 * NH + o] = c1v;  // c1[None]
        }
      }
    }
  }
}

// ---------------------------------------------------------------------------
extern "C" void kernel_launch(void* const* d_in, const int* in_sizes, int n_in,
                              void* d_out, int out_size, void* d_ws, size_t ws_size,
                              hipStream_t stream) {
  const float*      x       = (const float*)d_in[0];
  const long long*  ei      = (const long long*)d_in[1];   // int64 edge_index [2,E]
  const float*      h       = (const float*)d_in[2];       // [1,N,H]
  const float*      c       = (const float*)d_in[3];       // [1,N,H]
  const float*      W       = (const float*)d_in[4];
  const float*      att_src = (const float*)d_in[5];
  const float*      att_dst = (const float*)d_in[6];
  const float*      bias    = (const float*)d_in[7];
  const float*      Wih     = (const float*)d_in[8];
  const float*      Whh     = (const float*)d_in[9];
  float*            out     = (float*)d_out;

  const int n = in_sizes[0] / DD;      // 50000
  const int E = in_sizes[1] / 2;       // 600000
  const int Etot = E + n;              // with self loops

  // workspace layout (floats, all 32B-aligned by construction)
  float* ws    = (float*)d_ws;
  float* xs    = ws;                        // n*128
  float* accum = xs    + (size_t)n * DD;    // n*128
  float* a_src = accum + (size_t)n * DD;    // n
  float* a_dst = a_src + n;                 // n
  float* emax  = a_dst + n;                 // n
  float* esum  = emax  + n;                 // n
  float* ev    = esum  + n;                 // E+n
  unsigned short* packW  = (unsigned short*)(ev + Etot);          // 128*128 bf16
  unsigned short* packIh = packW  + (size_t)DD * DD;              // 128*512 bf16
  unsigned short* packHh = packIh + (size_t)DD * 4 * DD;          // 128*512 bf16
  (void)ws_size; (void)n_in; (void)out_size;

  const int waveTiles  = (n + 31) / 32;        // 32 rows per wave
  const int gemmBlocks = (waveTiles + 7) / 8;  // 8 waves per block

  // weight fragment packing (tiny, once per launch; L2-resident afterwards)
  k_pack_b<<<(4 * (DD / 16) * 32 + 255) / 256, 256, 0, stream>>>(W, packW, DD, 0);
  k_pack_b<<<(4 * (4 * DD / 16) * 32 + 255) / 256, 256, 0, stream>>>(Wih, packIh, 4 * DD, 1);
  k_pack_b<<<(4 * (4 * DD / 16) * 32 + 255) / 256, 256, 0, stream>>>(Whh, packHh, 4 * DD, 1);

  k_gemm_xs<<<gemmBlocks, 256, 0, stream>>>(x, packW, xs, n);
  k_att<<<(n + 7) / 8, 256, 0, stream>>>(xs, att_src, att_dst, a_src, a_dst, n);
  k_init<<<(n * DD + 255) / 256, 256, 0, stream>>>(accum, emax, esum, n);
  k_edge_max<<<(Etot + 255) / 256, 256, 0, stream>>>(ei, a_src, a_dst, ev, emax, Etot, E);
  k_edge_scatter<<<(Etot + 7) / 8, 256, 0, stream>>>(ei, ev, emax, xs, esum, accum, Etot, E);
  k_lstm<<<gemmBlocks, 256, 0, stream>>>(accum, esum, bias, h, c, packIh, packHh, out, n);
}
